// GroupMultiHeadAttention_58480274703166
// MI455X (gfx1250) — matrix operand order repaired and checked by reading, not compiled, not run on hardware
//
#include <hip/hip_runtime.h>

#define Bc   2
#define Lc   2048
#define Dc   2048
#define NHc  32
#define KVGc 8
#define GSc  4      // heads per kv group
#define HDc  64

typedef __attribute__((ext_vector_type(16))) __bf16        v16bf;
typedef __attribute__((ext_vector_type(8)))  float         v8f;
typedef unsigned short u16t;
typedef __attribute__((ext_vector_type(8)))  u16t          ushort8;
typedef __attribute__((ext_vector_type(4)))  unsigned int  v4u;
typedef __attribute__((ext_vector_type(8)))  int           v8i;
typedef __attribute__((ext_vector_type(4)))  int           v4i;

union BFrag { v16bf v; ushort8 h[2]; u16t u[16]; };

__device__ __forceinline__ u16t f2bf(float f) {
    unsigned u = __float_as_uint(f);
    u += 0x7FFFu + ((u >> 16) & 1u);   // round-to-nearest-even
    return (u16t)(u >> 16);
}
__device__ __forceinline__ float bf2f(u16t h) {
    return __uint_as_float(((unsigned)h) << 16);
}

// low 32 bits of a generic pointer into LDS == LDS byte offset (aperture translation)
__device__ __forceinline__ unsigned lds_off(const void* p) {
    return (unsigned)(size_t)p;
}

// ---- Tensor Data Mover: 2-D bf16 tile load (tile fully in-bounds; tensor dims == tile dims)
// tw = contiguous elements per row (dim0), th = rows (dim1), stride = row stride in elements
__device__ __forceinline__ void tdm_load_2d_bf16(unsigned ldsoff, const void* gptr,
                                                 unsigned tw, unsigned th,
                                                 unsigned long long stride) {
    unsigned long long ga = (unsigned long long)(size_t)gptr;
    v4u g0;
    g0[0] = 1u;                                           // count=1 (valid descriptor)
    g0[1] = ldsoff;                                       // lds_addr
    g0[2] = (unsigned)ga;                                 // global_addr[31:0]
    g0[3] = (unsigned)((ga >> 32) & 0x01FFFFFFull)        // global_addr[56:32]
          | 0x80000000u;                                  // type=2 ("image")
    v8i g1;
    g1[0] = (int)(1u << 16);                              // data_size=1 (2 bytes); wg_mask=0
    g1[1] = (int)((tw & 0xFFFFu) << 16);                  // tensor_dim0[15:0] @ bits[63:48]
    g1[2] = (int)(((tw >> 16) & 0xFFFFu)                  // tensor_dim0[31:16]
          | ((th & 0xFFFFu) << 16));                      // tensor_dim1[15:0]
    g1[3] = (int)(((th >> 16) & 0xFFFFu)                  // tensor_dim1[31:16]
          | ((tw & 0xFFFFu) << 16));                      // tile_dim0
    g1[4] = (int)(th & 0xFFFFu);                          // tile_dim1 (tile_dim2=0)
    g1[5] = (int)(unsigned)(stride & 0xFFFFFFFFull);      // tensor_dim0_stride[31:0]
    g1[6] = (int)(unsigned)((stride >> 32) & 0xFFFFull);  // stride[47:32]; dim1_stride lo=0
    g1[7] = 0;
    v4i z4 = {0, 0, 0, 0};
#if defined(__clang_major__) && (__clang_major__ >= 23)
    v8i z8 = {0, 0, 0, 0, 0, 0, 0, 0};
    __builtin_amdgcn_tensor_load_to_lds(g0, g1, z4, z4, z8, 0);
#else
    __builtin_amdgcn_tensor_load_to_lds(g0, g1, z4, z4, 0);
#endif
}

// ---------------- elementwise converters ----------------

__global__ void cvt_f32_to_bf16(const float* __restrict__ in, u16t* __restrict__ out, int n) {
    int i = blockIdx.x * blockDim.x + threadIdx.x;
    if (i < n) out[i] = f2bf(in[i]);
}

// W [K,N] f32 -> Wt [N,K] bf16
__global__ void wtrans_bf16(const float* __restrict__ W, u16t* __restrict__ Wt, int K, int N) {
    int i = blockIdx.x * blockDim.x + threadIdx.x;
    if (i >= K * N) return;
    int k = i / N, n = i - k * N;
    Wt[(size_t)n * K + k] = f2bf(W[i]);
}

// ---------------- WMMA GEMM: C[M,N] = A[M,Kd] @ Bt[N,Kd]^T ----------------
// 4-wave workgroup computes a 64x64 tile; TDM double-buffers A/B 32-K chunks in LDS.
template <bool WRITE_F32>
__global__ __launch_bounds__(128)
void gemm_wmma(const u16t* __restrict__ A, const u16t* __restrict__ Bt,
               void* __restrict__ C, const float* __restrict__ bias,
               int M, int N, int Kd) {
    __shared__ __align__(16) u16t Alds[2][64 * 32];
    __shared__ __align__(16) u16t Blds[2][64 * 32];

    const int tid  = threadIdx.x;
    const int wave = tid >> 5;
    const int lane = tid & 31;
    const int nh   = lane & 15;
    const int hi   = lane >> 4;
    const int m0   = blockIdx.x * 64;
    const int n0   = blockIdx.y * 64;

    v8f acc[4] = {};
    const int nchunks = Kd / 32;

    if (wave == 0) {
        tdm_load_2d_bf16(lds_off(&Alds[0][0]), A + (size_t)m0 * Kd, 32, 64, (unsigned long long)Kd);
        tdm_load_2d_bf16(lds_off(&Blds[0][0]), Bt + (size_t)n0 * Kd, 32, 64, (unsigned long long)Kd);
    }

    for (int ci = 0; ci < nchunks; ++ci) {
        const int buf = ci & 1;
        if (wave == 0) {
            if (ci + 1 < nchunks) {
                const int kc = (ci + 1) * 32;
                tdm_load_2d_bf16(lds_off(&Alds[buf ^ 1][0]), A + (size_t)m0 * Kd + kc, 32, 64,
                                 (unsigned long long)Kd);
                tdm_load_2d_bf16(lds_off(&Blds[buf ^ 1][0]), Bt + (size_t)n0 * Kd + kc, 32, 64,
                                 (unsigned long long)Kd);
                __builtin_amdgcn_s_wait_tensorcnt(2);   // current chunk landed; next in flight
            } else {
                __builtin_amdgcn_s_wait_tensorcnt(0);
            }
        }
        __syncthreads();   // chunk `buf` visible to all waves

        // gather ALL fragments first -> one ds clause, one wait, then WMMA burst
        BFrag a, b[4];
        const u16t* arow = &Alds[buf][(wave * 16 + nh) * 32];
        a.h[0] = *(const ushort8*)(arow + hi * 8);
        a.h[1] = *(const ushort8*)(arow + 16 + hi * 8);
#pragma unroll
        for (int t = 0; t < 4; ++t) {
            const u16t* brow = &Blds[buf][(t * 16 + nh) * 32 + hi * 8];
            b[t].h[0] = *(const ushort8*)(brow);
            b[t].h[1] = *(const ushort8*)(brow + 16);
        }
#pragma unroll
        for (int t = 0; t < 4; ++t) {
            acc[t] = __builtin_amdgcn_wmma_f32_16x16x32_bf16(
                false, a.v, false, b[t].v, (short)0, acc[t], false, false);
        }
        __syncthreads();   // all reads of `buf` done before it is re-filled
    }

#pragma unroll
    for (int t = 0; t < 4; ++t) {
        int n = n0 + t * 16 + nh;
#pragma unroll
        for (int r = 0; r < 8; ++r) {
            int m = m0 + wave * 16 + r + hi * 8;
            float v = acc[t][r];
            if (WRITE_F32) ((float*)C)[(size_t)m * N + n] = v + bias[n];
            else           ((u16t*)C)[(size_t)m * N + n] = f2bf(v);
        }
    }
}

// ---------------- RoPE + reshape: [B,L,NH*hd] bf16 -> [B,NH,L,hd] bf16 ----------------
__global__ void rope_reshape(const u16t* __restrict__ T, const float* __restrict__ cosT,
                             const float* __restrict__ sinT, u16t* __restrict__ Out, int NH) {
    long long idx = (long long)blockIdx.x * blockDim.x + threadIdx.x;
    if (idx >= (long long)Bc * Lc * NH * HDc) return;
    int d = idx & (HDc - 1);
    long long t = idx >> 6;
    int h = (int)(t % NH); t /= NH;
    int l = (int)(t % Lc);
    int b = (int)(t / Lc);
    const u16t* row = T + ((size_t)(b * Lc + l) * NH + h) * HDc;
    float v   = bf2f(row[d]);
    float rot = (d < HDc / 2) ? -bf2f(row[d + HDc / 2]) : bf2f(row[d - HDc / 2]);
    float o   = v * cosT[l * HDc + d] + rot * sinT[l * HDc + d];
    Out[(((size_t)b * NH + h) * Lc + l) * HDc + d] = f2bf(o);
}

// V: [B,L,G*hd] bf16 -> [B,G,hd,L] bf16 (transposed for contiguous B-fragments)
__global__ void v_transpose(const u16t* __restrict__ Vin, u16t* __restrict__ Out) {
    long long idx = (long long)blockIdx.x * blockDim.x + threadIdx.x;
    if (idx >= (long long)Bc * Lc * KVGc * HDc) return;
    int d = idx & (HDc - 1);
    long long t = idx >> 6;
    int g = (int)(t % KVGc); t /= KVGc;
    int l = (int)(t % Lc);
    int b = (int)(t / Lc);
    Out[(((size_t)b * KVGc + g) * HDc + d) * Lc + l] =
        Vin[((size_t)(b * Lc + l) * KVGc + g) * HDc + d];
}

// ---------------- flash attention: 1 wave per (b, h, 16-row query block) ----------------
// TDM double-buffers the 32-key K tile (32x64) and transposed V tile (64x32) in LDS.
__global__ __launch_bounds__(32)
void attn_wmma(const u16t* __restrict__ Qh, const u16t* __restrict__ Kh,
               const u16t* __restrict__ VhT, u16t* __restrict__ Ob) {
    __shared__ __align__(16) u16t Klds[2][32 * HDc];   // [key][d]
    __shared__ __align__(16) u16t Vlds[2][HDc * 32];   // [d][key]
    __shared__ __align__(16) u16t P_lds[16 * 32];

    const int lane = threadIdx.x;
    const int nh   = lane & 15;
    const int hi   = lane >> 4;
    const int qb   = blockIdx.x;
    const int h    = blockIdx.y;
    const int b    = blockIdx.z;
    const int g    = h / GSc;
    const int q0   = qb * 16;

    const u16t* Qbase = Qh  + (((size_t)b * NHc + h) * Lc + q0) * HDc;
    const u16t* Kbase = Kh  + (((size_t)b * KVGc + g) * Lc) * HDc;
    const u16t* Vbase = VhT + (((size_t)b * KVGc + g) * HDc) * (size_t)Lc;

    // Q A-fragments (hd=64 -> two K=32 chunks), loaded once
    BFrag qf[2];
#pragma unroll
    for (int c = 0; c < 2; ++c) {
        const u16t* qrow = Qbase + (size_t)nh * HDc + c * 32;
        qf[c].h[0] = *(const ushort8*)(qrow + hi * 8);
        qf[c].h[1] = *(const ushort8*)(qrow + 16 + hi * 8);
    }

    float m_i[8], l_i[8];
    v8f oacc[4] = {};
#pragma unroll
    for (int r = 0; r < 8; ++r) { m_i[r] = -__builtin_inff(); l_i[r] = 0.0f; }

    const float scale = 0.125f;               // 1/sqrt(64)
    const int kend    = q0 + 16;              // causal: last needed key = q0+15
    const int nchunks = (kend + 31) / 32;

    // prologue DMA: chunk 0
    tdm_load_2d_bf16(lds_off(&Klds[0][0]), Kbase, HDc, 32, (unsigned long long)HDc);
    tdm_load_2d_bf16(lds_off(&Vlds[0][0]), Vbase, 32, HDc, (unsigned long long)Lc);

    for (int ci = 0; ci < nchunks; ++ci) {
        const int k0  = ci * 32;
        const int buf = ci & 1;
        if (ci + 1 < nchunks) {
            const int kn = (ci + 1) * 32;
            tdm_load_2d_bf16(lds_off(&Klds[buf ^ 1][0]), Kbase + (size_t)kn * HDc, HDc, 32,
                             (unsigned long long)HDc);
            tdm_load_2d_bf16(lds_off(&Vlds[buf ^ 1][0]), Vbase + kn, 32, HDc,
                             (unsigned long long)Lc);
            __builtin_amdgcn_s_wait_tensorcnt(2);
        } else {
            __builtin_amdgcn_s_wait_tensorcnt(0);
        }

        // gather all 4 K fragments, then the 4 S WMMAs back-to-back
        BFrag kf[4];   // [c][s] : c = k-dim chunk, s = key half
#pragma unroll
        for (int c = 0; c < 2; ++c) {
            const u16t* kr0 = &Klds[buf][nh * HDc + c * 32 + hi * 8];
            kf[c * 2 + 0].h[0] = *(const ushort8*)(kr0);
            kf[c * 2 + 0].h[1] = *(const ushort8*)(kr0 + 16);
            const u16t* kr1 = &Klds[buf][(16 + nh) * HDc + c * 32 + hi * 8];
            kf[c * 2 + 1].h[0] = *(const ushort8*)(kr1);
            kf[c * 2 + 1].h[1] = *(const ushort8*)(kr1 + 16);
        }
        v8f s0 = {}, s1 = {};
        s0 = __builtin_amdgcn_wmma_f32_16x16x32_bf16(false, qf[0].v, false, kf[0].v,
                                                     (short)0, s0, false, false);
        s1 = __builtin_amdgcn_wmma_f32_16x16x32_bf16(false, qf[0].v, false, kf[1].v,
                                                     (short)0, s1, false, false);
        s0 = __builtin_amdgcn_wmma_f32_16x16x32_bf16(false, qf[1].v, false, kf[2].v,
                                                     (short)0, s0, false, false);
        s1 = __builtin_amdgcn_wmma_f32_16x16x32_bf16(false, qf[1].v, false, kf[3].v,
                                                     (short)0, s1, false, false);

        // online softmax, per query row (8 rows per half-wave)
#pragma unroll
        for (int r = 0; r < 8; ++r) {
            int m    = r + hi * 8;
            int qpos = q0 + m;
            float v0 = (k0 + nh      <= qpos) ? s0[r] * scale : -__builtin_inff();
            float v1 = (k0 + 16 + nh <= qpos) ? s1[r] * scale : -__builtin_inff();
            float mx = fmaxf(v0, v1);
#pragma unroll
            for (int off = 1; off < 16; off <<= 1)
                mx = fmaxf(mx, __shfl_xor(mx, off));
            float newm  = fmaxf(m_i[r], mx);
            float alpha = __expf(m_i[r] - newm);
            m_i[r] = newm;
            float p0 = (v0 > -__builtin_inff()) ? __expf(v0 - newm) : 0.0f;
            float p1 = (v1 > -__builtin_inff()) ? __expf(v1 - newm) : 0.0f;
            float rs = p0 + p1;
#pragma unroll
            for (int off = 1; off < 16; off <<= 1)
                rs += __shfl_xor(rs, off);
            l_i[r] = l_i[r] * alpha + rs;
#pragma unroll
            for (int t = 0; t < 4; ++t) oacc[t][r] = oacc[t][r] * alpha;
            P_lds[m * 32 + nh]      = f2bf(p0);
            P_lds[m * 32 + 16 + nh] = f2bf(p1);
        }
        __syncthreads();

        // gather P A-fragment + all 4 V fragments, then the 4 O WMMAs back-to-back
        BFrag pf, vf[4];
        pf.h[0] = *(const ushort8*)(&P_lds[nh * 32 + hi * 8]);
        pf.h[1] = *(const ushort8*)(&P_lds[nh * 32 + 16 + hi * 8]);
#pragma unroll
        for (int dt = 0; dt < 4; ++dt) {
            const u16t* vrow = &Vlds[buf][(dt * 16 + nh) * 32 + hi * 8];
            vf[dt].h[0] = *(const ushort8*)(vrow);
            vf[dt].h[1] = *(const ushort8*)(vrow + 16);
        }
#pragma unroll
        for (int dt = 0; dt < 4; ++dt) {
            oacc[dt] = __builtin_amdgcn_wmma_f32_16x16x32_bf16(false, pf.v, false, vf[dt].v,
                                                               (short)0, oacc[dt], false, false);
        }
        __syncthreads();
    }

    // normalize and write Ob[b, l, h*hd + d] (bf16)
#pragma unroll
    for (int dt = 0; dt < 4; ++dt) {
#pragma unroll
        for (int r = 0; r < 8; ++r) {
            int m = r + hi * 8;
            float o = oacc[dt][r] / l_i[r];
            Ob[((size_t)(b * Lc + q0 + m)) * Dc + h * HDc + dt * 16 + nh] = f2bf(o);
        }
    }
}

// ---------------- launcher ----------------

extern "C" void kernel_launch(void* const* d_in, const int* in_sizes, int n_in,
                              void* d_out, int out_size, void* d_ws, size_t ws_size,
                              hipStream_t stream) {
    const float* x    = (const float*)d_in[0];
    // d_in[1] = mask (ignored; causal mask applied analytically)
    const float* cosT = (const float*)d_in[2];
    const float* sinT = (const float*)d_in[3];
    const float* Wq   = (const float*)d_in[4];
    const float* Wk   = (const float*)d_in[5];
    const float* Wv   = (const float*)d_in[6];
    const float* Wo   = (const float*)d_in[7];
    const float* bo   = (const float*)d_in[8];
    float* out = (float*)d_out;

    const int M  = Bc * Lc;          // 4096
    const int Nk = KVGc * HDc;       // 512

    char* p = (char*)d_ws;
    auto take = [&](size_t elems) { u16t* r = (u16t*)p; p += elems * sizeof(u16t); return r; };
    u16t* Xb   = take((size_t)M * Dc);
    u16t* WqT  = take((size_t)Dc * Dc);
    u16t* WkT  = take((size_t)Nk * Dc);
    u16t* WvT  = take((size_t)Nk * Dc);
    u16t* WoT  = take((size_t)Dc * Dc);
    u16t* Qtmp = take((size_t)M * Dc);
    u16t* Ktmp = take((size_t)M * Nk);
    u16t* Vtmp = take((size_t)M * Nk);
    u16t* Qh   = take((size_t)Bc * NHc * Lc * HDc);
    u16t* Kh   = take((size_t)Bc * KVGc * Lc * HDc);
    u16t* VhT  = take((size_t)Bc * KVGc * HDc * Lc);
    u16t* Ob   = take((size_t)M * Dc);
    (void)ws_size; (void)in_sizes; (void)n_in; (void)out_size;

    const int TB = 256;
    cvt_f32_to_bf16<<<(M * Dc + TB - 1) / TB, TB, 0, stream>>>(x, Xb, M * Dc);
    wtrans_bf16<<<(Dc * Dc + TB - 1) / TB, TB, 0, stream>>>(Wq, WqT, Dc, Dc);
    wtrans_bf16<<<(Dc * Nk + TB - 1) / TB, TB, 0, stream>>>(Wk, WkT, Dc, Nk);
    wtrans_bf16<<<(Dc * Nk + TB - 1) / TB, TB, 0, stream>>>(Wv, WvT, Dc, Nk);
    wtrans_bf16<<<(Dc * Dc + TB - 1) / TB, TB, 0, stream>>>(Wo, WoT, Dc, Dc);

    gemm_wmma<false><<<dim3(M / 64, Dc / 64), 128, 0, stream>>>(Xb, WqT, Qtmp, nullptr, M, Dc, Dc);
    gemm_wmma<false><<<dim3(M / 64, Nk / 64), 128, 0, stream>>>(Xb, WkT, Ktmp, nullptr, M, Nk, Dc);
    gemm_wmma<false><<<dim3(M / 64, Nk / 64), 128, 0, stream>>>(Xb, WvT, Vtmp, nullptr, M, Nk, Dc);

    rope_reshape<<<(M * Dc + TB - 1) / TB, TB, 0, stream>>>(Qtmp, cosT, sinT, Qh, NHc);
    rope_reshape<<<(M * Nk + TB - 1) / TB, TB, 0, stream>>>(Ktmp, cosT, sinT, Kh, KVGc);
    v_transpose<<<(M * Nk + TB - 1) / TB, TB, 0, stream>>>(Vtmp, VhT);

    attn_wmma<<<dim3(Lc / 16, NHc, Bc), 32, 0, stream>>>(Qh, Kh, VhT, Ob);

    gemm_wmma<true><<<dim3(M / 64, Dc / 64), 128, 0, stream>>>(Ob, WoT, out, bo, M, Dc, Dc);
}
